// CausalSelfAttention_78743930405118
// MI455X (gfx1250) — matrix operand order repaired and checked by reading, not compiled, not run on hardware
//
#include <hip/hip_runtime.h>
#include <hip/hip_bf16.h>

// ---------------------------------------------------------------------------
// CausalSelfAttention on gfx1250 (MI455X).
// bf16 WMMA GEMMs + flash attention; double-buffered TDM tile staging;
// transposed weight/V layouts so every WMMA fragment is a contiguous
// per-lane 32B run (2x b128 loads, no lane gathers).
// B=4, T=2048, C=1024, H=16, D=64.
// ---------------------------------------------------------------------------

typedef __bf16 bf16_t;
typedef __attribute__((ext_vector_type(16))) __bf16 v16bf;
typedef __attribute__((ext_vector_type(8)))  __bf16 v8bf;
typedef __attribute__((ext_vector_type(8)))  float  v8f;
typedef __attribute__((ext_vector_type(4)))  unsigned int v4u;
typedef __attribute__((ext_vector_type(8)))  int v8i;
typedef __attribute__((ext_vector_type(4)))  int v4i;

#define B_  4
#define T_  2048
#define C_  1024
#define H_  16
#define D_  64

static __device__ __forceinline__ bf16_t f2bf(float f) { return (bf16_t)f; }

static __device__ __forceinline__ v8f zero_v8f() {
    v8f z;
#pragma unroll
    for (int e = 0; e < 8; ++e) z[e] = 0.0f;
    return z;
}

static __device__ __forceinline__ v16bf concat8(v8bf lo, v8bf hi) {
    return __builtin_shufflevector(lo, hi, 0, 1, 2, 3, 4, 5, 6, 7,
                                           8, 9, 10, 11, 12, 13, 14, 15);
}

static __device__ __forceinline__ v8f wmma_bf16(v16bf a, v16bf b, v8f c) {
    return __builtin_amdgcn_wmma_f32_16x16x32_bf16(
        false, a, false, b, (short)0, c, false, false);
}

// Low 32 bits of a generic pointer to __shared__ = LDS byte offset.
static __device__ __forceinline__ unsigned lds_offset(const void* p) {
    return (unsigned)(unsigned long long)p;
}

// ---------------------------------------------------------------------------
// TDM: async 2D bf16 tile load Global -> LDS (D# per CDNA5 ISA).
// ---------------------------------------------------------------------------
static __device__ __forceinline__ void tdm_load_2d_bf16(
    unsigned lds_off, const void* gptr,
    unsigned tile_d0, unsigned tile_d1,
    unsigned tensor_d0, unsigned tensor_d1,
    unsigned long long stride0_elems)
{
    unsigned long long ga = (unsigned long long)gptr;
    v4u g0;
    g0[0] = 1u;                                             // count=1, user mode
    g0[1] = lds_off;                                        // lds_addr (bytes)
    g0[2] = (unsigned)ga;                                   // global_addr[31:0]
    g0[3] = (unsigned)((ga >> 32) & 0x01FFFFFFu) | (2u << 30); // addr[56:32]|type=2

    v8i g1;
    g1[0] = (int)(1u << 16);                                // mask=0, data_size=1 (2B)
    g1[1] = (int)((tensor_d0 & 0xFFFFu) << 16);             // tensor_dim0[15:0]
    g1[2] = (int)(((tensor_d0 >> 16) & 0xFFFFu) | ((tensor_d1 & 0xFFFFu) << 16));
    g1[3] = (int)(((tensor_d1 >> 16) & 0xFFFFu) | ((tile_d0 & 0xFFFFu) << 16));
    g1[4] = (int)(tile_d1 & 0xFFFFu);                       // tile_dim1, tile_dim2=0
    g1[5] = (int)(unsigned)(stride0_elems & 0xFFFFFFFFull);
    g1[6] = (int)(unsigned)((stride0_elems >> 32) & 0xFFFFull);
    g1[7] = 0;

    v4i gz4;
#pragma unroll
    for (int i = 0; i < 4; ++i) gz4[i] = 0;
#if defined(__clang_major__) && __clang_major__ >= 23
    v8i gz8;
#pragma unroll
    for (int i = 0; i < 8; ++i) gz8[i] = 0;
    __builtin_amdgcn_tensor_load_to_lds(g0, g1, gz4, gz4, gz8, 0);
#else
    __builtin_amdgcn_tensor_load_to_lds(g0, g1, gz4, gz4, 0);
#endif
}

// ---------------------------------------------------------------------------
// Fragment builders (k = kh*16 + e contiguous, same order in A and B).
// All reads are contiguous 2x16B per lane.
// ---------------------------------------------------------------------------

// A-fragment (16x32 bf16, row-major): lane = kh*16+m;
// K = kh*16 + e contiguous
static __device__ __forceinline__ v16bf load_a_frag(const bf16_t* __restrict__ A,
                                                    int lda, int m0, int k0, int lane) {
    int m  = m0 + (lane & 15);
    int kh = lane >> 4;
    const bf16_t* p = A + (size_t)m * lda + k0 + kh * 16;
    return concat8(*(const v8bf*)p, *(const v8bf*)(p + 8));
}

// B-fragment from an N-major (transposed) 16x32 LDS tile: lane = kh*16+n
// holds column n of B = row n of tile; K = kh*16 + e contiguous.
static __device__ __forceinline__ v16bf b_frag_from_ldsT(const bf16_t* wtT, int lane) {
    int n  = lane & 15;
    int kh = lane >> 4;
    const v8bf* p = (const v8bf*)(wtT + n * 32 + kh * 16);
    return concat8(p[0], p[1]);
}

// B-fragment of K^T from a 32x64 LDS tile (key,d): column n = key -> contiguous d
static __device__ __forceinline__ v16bf bT_frag_from_lds(const bf16_t* kt,
                                                         int key0, int d0, int lane) {
    int key = key0 + (lane & 15);
    int kh  = lane >> 4;
    const v8bf* p = (const v8bf*)(kt + key * D_ + d0 + kh * 16);
    return concat8(p[0], p[1]);
}

// B-fragment of V from a transposed 64x32 LDS tile (d,key):
// lane holds column d = col0+n; K = key = kh*16 + e contiguous.
static __device__ __forceinline__ v16bf v_frag_from_ldsT(const bf16_t* vtT,
                                                         int col0, int lane) {
    int nn = col0 + (lane & 15);
    int kh = lane >> 4;
    const v8bf* p = (const v8bf*)(vtT + nn * 32 + kh * 16);
    return concat8(p[0], p[1]);
}

// ---------------------------------------------------------------------------
// Kernel 1a: fp32 -> bf16 conversion (grid-stride)
// ---------------------------------------------------------------------------
__global__ void __launch_bounds__(256) cvt_bf16_kernel(const float* __restrict__ in,
                                                       bf16_t* __restrict__ out,
                                                       unsigned int n) {
    unsigned int i      = blockIdx.x * blockDim.x + threadIdx.x;
    unsigned int stride = gridDim.x * blockDim.x;
    for (; i < n; i += stride) out[i] = f2bf(in[i]);
}

// ---------------------------------------------------------------------------
// Kernel 1b: fp32 (KxN) -> bf16 transposed (NxK), LDS-tiled for coalescing.
// Block (32,8); 32x32 tile.
// ---------------------------------------------------------------------------
__global__ void __launch_bounds__(256) cvt_transpose_kernel(const float* __restrict__ in,
                                                            bf16_t* __restrict__ outT,
                                                            int K, int N) {
    __shared__ float tile[32][33];
    const int k0 = blockIdx.y * 32;
    const int n0 = blockIdx.x * 32;
#pragma unroll
    for (int i = 0; i < 4; ++i)
        tile[threadIdx.y + i * 8][threadIdx.x] =
            in[(size_t)(k0 + threadIdx.y + i * 8) * N + n0 + threadIdx.x];
    __syncthreads();
#pragma unroll
    for (int i = 0; i < 4; ++i)
        outT[(size_t)(n0 + threadIdx.y + i * 8) * K + k0 + threadIdx.x] =
            f2bf(tile[threadIdx.x][threadIdx.y + i * 8]);
}

// ---------------------------------------------------------------------------
// Kernel 2: QKV = Xb(8192x1024) @ W (W stored N-major: WaT[3072][1024]).
// 4 waves/block stacked in M share double-buffered TDM-staged W strips.
// Q,K stored (B,H,T,D); V stored transposed (B,H,D,T).
// ---------------------------------------------------------------------------
__global__ void __launch_bounds__(128) gemm_qkv_kernel(const bf16_t* __restrict__ Xb,
                                                       const bf16_t* __restrict__ WaT,
                                                       bf16_t* __restrict__ Q,
                                                       bf16_t* __restrict__ K,
                                                       bf16_t* __restrict__ Vt) {
    __shared__ __align__(16) bf16_t wtT[2][16 * 32];
    const int lane = threadIdx.x & 31;
    const int wave = threadIdx.x >> 5;
    const int n0 = blockIdx.x * 16;
    const int m0 = blockIdx.y * 64 + wave * 16;

    if (wave == 0)  // preload first strip: rows n0..n0+15, cols 0..31 of WaT
        tdm_load_2d_bf16(lds_offset(wtT[0]), WaT + (size_t)n0 * C_,
                         32, 16, (unsigned)C_, (unsigned)(3 * C_ - n0),
                         (unsigned long long)C_);

    v8f acc = zero_v8f();
    for (int k0 = 0; k0 < C_; k0 += 32) {
        const int cur = (k0 >> 5) & 1;
        if (wave == 0) {
            if (k0 + 32 < C_) {
                tdm_load_2d_bf16(lds_offset(wtT[cur ^ 1]),
                                 WaT + (size_t)n0 * C_ + k0 + 32,
                                 32, 16, (unsigned)(C_ - k0 - 32),
                                 (unsigned)(3 * C_ - n0), (unsigned long long)C_);
                __builtin_amdgcn_s_wait_tensorcnt(1);   // current strip done
            } else {
                __builtin_amdgcn_s_wait_tensorcnt(0);
            }
        }
        __syncthreads();
        if (k0 + 32 < C_)
            __builtin_prefetch(Xb + (size_t)(m0 + (lane & 15)) * C_ + k0 + 32, 0, 3);
        v16bf a = load_a_frag(Xb, C_, m0, k0, lane);
        v16bf b = b_frag_from_ldsT(wtT[cur], lane);
        acc = wmma_bf16(a, b, acc);
        __syncthreads();
    }

    const int n   = n0 + (lane & 15);
    const int sec = n >> 10;          // 0=Q 1=K 2=V
    const int c   = n & (C_ - 1);
    const int h   = c >> 6;
    const int d   = c & (D_ - 1);
    if (sec == 2) {                   // V transposed: Vt[b,h,d,t]
#pragma unroll
        for (int v = 0; v < 8; ++v) {
            int m  = m0 + v + ((lane >> 4) << 3);
            int b_ = m >> 11;
            int t  = m & (T_ - 1);
            Vt[(((size_t)(b_ * H_ + h)) * D_ + d) * T_ + t] = f2bf(acc[v]);
        }
    } else {
        bf16_t* dst = (sec == 0) ? Q : K;
#pragma unroll
        for (int v = 0; v < 8; ++v) {
            int m  = m0 + v + ((lane >> 4) << 3);
            int b_ = m >> 11;
            int t  = m & (T_ - 1);
            dst[(((size_t)(b_ * H_ + h)) * T_ + t) * D_ + d] = f2bf(acc[v]);
        }
    }
}

// ---------------------------------------------------------------------------
// Kernel 3: flash attention. Block = 64 query rows (4 waves x 16).
// K (32x64) and Vt (64x32) tiles double-buffered in LDS via TDM.
// ---------------------------------------------------------------------------
__global__ void __launch_bounds__(128) flash_attn_kernel(const bf16_t* __restrict__ Q,
                                                         const bf16_t* __restrict__ K,
                                                         const bf16_t* __restrict__ Vt,
                                                         bf16_t* __restrict__ Y) {
    __shared__ __align__(16) bf16_t kt[2][32 * D_];     // 8 KiB
    __shared__ __align__(16) bf16_t vtT[2][D_ * 32];    // 8 KiB
    __shared__ __align__(16) bf16_t pbuf[4][16 * 32];   // per-wave P staging

    const int lane = threadIdx.x & 31;
    const int wave = threadIdx.x >> 5;
    const int q0b = blockIdx.x * 64;
    const int q0  = q0b + wave * 16;
    const int bh  = blockIdx.y;
    const int b   = bh >> 4;
    const int h   = bh & (H_ - 1);

    const bf16_t* Qp  = Q  + (size_t)bh * T_ * D_;
    const bf16_t* Kp  = K  + (size_t)bh * T_ * D_;
    const bf16_t* Vtp = Vt + (size_t)bh * D_ * T_;

    const v16bf aq0 = load_a_frag(Qp, D_, q0, 0, lane);
    const v16bf aq1 = load_a_frag(Qp, D_, q0, 32, lane);

    v8f o0 = zero_v8f(), o1 = zero_v8f(), o2 = zero_v8f(), o3 = zero_v8f();
    float mrow[8], lrow[8];
#pragma unroll
    for (int v = 0; v < 8; ++v) { mrow[v] = -1e30f; lrow[v] = 0.0f; }

    const float SCALE = 0.125f;                  // 1/sqrt(64)
    const float LOG2E = 1.4426950408889634f;
    const int n  = lane & 15;
    const int rh = (lane >> 4) << 3;
    const int nblk = (q0b + 64 + 31) >> 5;       // causal key blocks for this block

    if (wave == 0) {                             // preload block 0
        tdm_load_2d_bf16(lds_offset(kt[0]), Kp,
                         D_, 32, D_, (unsigned)T_, (unsigned long long)D_);
        tdm_load_2d_bf16(lds_offset(vtT[0]), Vtp,
                         32, D_, (unsigned)T_, D_, (unsigned long long)T_);
    }

    for (int j = 0; j < nblk; ++j) {
        const int kb  = j * 32;
        const int cur = j & 1;
        if (wave == 0) {
            if (j + 1 < nblk) {                  // prefetch next K/V pair
                tdm_load_2d_bf16(lds_offset(kt[cur ^ 1]), Kp + (size_t)(kb + 32) * D_,
                                 D_, 32, D_, (unsigned)(T_ - kb - 32),
                                 (unsigned long long)D_);
                tdm_load_2d_bf16(lds_offset(vtT[cur ^ 1]), Vtp + kb + 32,
                                 32, D_, (unsigned)(T_ - kb - 32), D_,
                                 (unsigned long long)T_);
                __builtin_amdgcn_s_wait_tensorcnt(2);   // current pair done
            } else {
                __builtin_amdgcn_s_wait_tensorcnt(0);
            }
        }
        __syncthreads();

        if (kb < q0 + 16) {                      // wave-uniform causal skip
            const bf16_t* ktc  = kt[cur];
            const bf16_t* vtc  = vtT[cur];
            // S = (Q * K^T) * scale, two 16x16 tiles over 32 keys
            v8f s0 = zero_v8f(), s1 = zero_v8f();
            s0 = wmma_bf16(aq0, bT_frag_from_lds(ktc, 0,  0,  lane), s0);
            s0 = wmma_bf16(aq1, bT_frag_from_lds(ktc, 0,  32, lane), s0);
            s1 = wmma_bf16(aq0, bT_frag_from_lds(ktc, 16, 0,  lane), s1);
            s1 = wmma_bf16(aq1, bT_frag_from_lds(ktc, 16, 32, lane), s1);

            float mx[8];
#pragma unroll
            for (int v = 0; v < 8; ++v) {
                int qg = q0 + v + rh;
                float x0 = (kb + n      <= qg) ? s0[v] * SCALE : -1e30f;
                float x1 = (kb + 16 + n <= qg) ? s1[v] * SCALE : -1e30f;
                s0[v] = x0; s1[v] = x1;
                mx[v] = fmaxf(x0, x1);
            }
#pragma unroll
            for (int v = 0; v < 8; ++v) {
#pragma unroll
                for (int msk = 1; msk < 16; msk <<= 1)
                    mx[v] = fmaxf(mx[v], __shfl_xor(mx[v], msk));
            }

            float alpha[8], rs[8];
#pragma unroll
            for (int v = 0; v < 8; ++v) {
                float mn = fmaxf(mrow[v], mx[v]);
                alpha[v] = exp2f((mrow[v] - mn) * LOG2E);
                mrow[v]  = mn;
                float p0 = exp2f((s0[v] - mn) * LOG2E);
                float p1 = exp2f((s1[v] - mn) * LOG2E);
                s0[v] = p0; s1[v] = p1;
                rs[v] = p0 + p1;
            }
#pragma unroll
            for (int v = 0; v < 8; ++v) {
#pragma unroll
                for (int msk = 1; msk < 16; msk <<= 1)
                    rs[v] += __shfl_xor(rs[v], msk);
                lrow[v] = lrow[v] * alpha[v] + rs[v];
                o0[v] *= alpha[v]; o1[v] *= alpha[v];
                o2[v] *= alpha[v]; o3[v] *= alpha[v];
            }

            // Re-layout P (D-fragment -> A-fragment) through per-wave LDS
            bf16_t* pl = pbuf[wave];
#pragma unroll
            for (int v = 0; v < 8; ++v) {
                int m = v + rh;
                pl[m * 32 + n]      = f2bf(s0[v]);
                pl[m * 32 + 16 + n] = f2bf(s1[v]);
            }
            asm volatile("s_wait_dscnt 0" ::: "memory");   // wave-local LDS RAW fence
            v16bf pa;
            {
                int m  = lane & 15;
                int kh = lane >> 4;
                const v8bf* plo = (const v8bf*)(pl + m * 32 + kh * 16);
                const v8bf* phi = (const v8bf*)(pl + m * 32 + kh * 16 + 8);
                pa = concat8(*plo, *phi);
            }
            asm volatile("s_wait_dscnt 0" ::: "memory");

            // O += P(16x32) * V(32x64) from transposed V tile
            o0 = wmma_bf16(pa, v_frag_from_ldsT(vtc, 0,  lane), o0);
            o1 = wmma_bf16(pa, v_frag_from_ldsT(vtc, 16, lane), o1);
            o2 = wmma_bf16(pa, v_frag_from_ldsT(vtc, 32, lane), o2);
            o3 = wmma_bf16(pa, v_frag_from_ldsT(vtc, 48, lane), o3);
        }
        __syncthreads();                          // buffer reuse fence
    }

    // Normalize and store Y (B,T,C) bf16: y[b, t, h*64 + d]
#pragma unroll
    for (int v = 0; v < 8; ++v) {
        float inv = 1.0f / lrow[v];
        int t = q0 + v + rh;
        size_t base = ((size_t)(b * T_ + t)) * C_ + h * D_ + n;
        Y[base + 0 * 16] = f2bf(o0[v] * inv);
        Y[base + 1 * 16] = f2bf(o1[v] * inv);
        Y[base + 2 * 16] = f2bf(o2[v] * inv);
        Y[base + 3 * 16] = f2bf(o3[v] * inv);
    }
}

// ---------------------------------------------------------------------------
// Kernel 4: out(fp32) = Yb(8192x1024) @ Wp (stored N-major WpT[1024][1024])
// ---------------------------------------------------------------------------
__global__ void __launch_bounds__(128) gemm_out_kernel(const bf16_t* __restrict__ Yb,
                                                       const bf16_t* __restrict__ WpT,
                                                       float* __restrict__ out) {
    __shared__ __align__(16) bf16_t wtT[2][16 * 32];
    const int lane = threadIdx.x & 31;
    const int wave = threadIdx.x >> 5;
    const int n0 = blockIdx.x * 16;
    const int m0 = blockIdx.y * 64 + wave * 16;

    if (wave == 0)
        tdm_load_2d_bf16(lds_offset(wtT[0]), WpT + (size_t)n0 * C_,
                         32, 16, (unsigned)C_, (unsigned)(C_ - n0),
                         (unsigned long long)C_);

    v8f acc = zero_v8f();
    for (int k0 = 0; k0 < C_; k0 += 32) {
        const int cur = (k0 >> 5) & 1;
        if (wave == 0) {
            if (k0 + 32 < C_) {
                tdm_load_2d_bf16(lds_offset(wtT[cur ^ 1]),
                                 WpT + (size_t)n0 * C_ + k0 + 32,
                                 32, 16, (unsigned)(C_ - k0 - 32),
                                 (unsigned)(C_ - n0), (unsigned long long)C_);
                __builtin_amdgcn_s_wait_tensorcnt(1);
            } else {
                __builtin_amdgcn_s_wait_tensorcnt(0);
            }
        }
        __syncthreads();
        if (k0 + 32 < C_)
            __builtin_prefetch(Yb + (size_t)(m0 + (lane & 15)) * C_ + k0 + 32, 0, 3);
        v16bf a = load_a_frag(Yb, C_, m0, k0, lane);
        v16bf b = b_frag_from_ldsT(wtT[cur], lane);
        acc = wmma_bf16(a, b, acc);
        __syncthreads();
    }

    const int nn = n0 + (lane & 15);
#pragma unroll
    for (int v = 0; v < 8; ++v) {
        int m = m0 + v + ((lane >> 4) << 3);
        out[(size_t)m * C_ + nn] = acc[v];
    }
}

// ---------------------------------------------------------------------------
// Host-side launch
// ---------------------------------------------------------------------------
extern "C" void kernel_launch(void* const* d_in, const int* in_sizes, int n_in,
                              void* d_out, int out_size, void* d_ws, size_t ws_size,
                              hipStream_t stream) {
    const float* x  = (const float*)d_in[0];   // (B,T,C)
    const float* Wa = (const float*)d_in[1];   // (C,3C)
    const float* Wp = (const float*)d_in[2];   // (C,C)
    float* out = (float*)d_out;                // (B,T,C) fp32

    const size_t nX  = (size_t)B_ * T_ * C_;
    const size_t nWa = (size_t)C_ * 3 * C_;
    const size_t nWp = (size_t)C_ * C_;
    const size_t nQ  = (size_t)B_ * H_ * T_ * D_;

    char* ws = (char*)d_ws;
    size_t off = 0;
    bf16_t* Xb   = (bf16_t*)(ws + off); off += nX  * 2;  // x, row-major
    bf16_t* WaTb = (bf16_t*)(ws + off); off += nWa * 2;  // W_attn^T (3072x1024)
    bf16_t* WpTb = (bf16_t*)(ws + off); off += nWp * 2;  // W_proj^T (1024x1024)
    bf16_t* Qb   = (bf16_t*)(ws + off); off += nQ  * 2;  // (B,H,T,D)
    bf16_t* Kb   = (bf16_t*)(ws + off); off += nQ  * 2;  // (B,H,T,D)
    bf16_t* Vtb  = (bf16_t*)(ws + off); off += nQ  * 2;  // (B,H,D,T)
    bf16_t* Yb   = (bf16_t*)(ws + off); off += nX  * 2;  // (B,T,C)
    (void)ws_size; (void)in_sizes; (void)n_in; (void)out_size;

    cvt_bf16_kernel<<<1024, 256, 0, stream>>>(x, Xb, (unsigned int)nX);
    cvt_transpose_kernel<<<dim3(96, 32), dim3(32, 8), 0, stream>>>(Wa, WaTb, C_, 3 * C_);
    cvt_transpose_kernel<<<dim3(32, 32), dim3(32, 8), 0, stream>>>(Wp, WpTb, C_, C_);

    gemm_qkv_kernel<<<dim3(192, 128), 128, 0, stream>>>(Xb, WaTb, Qb, Kb, Vtb);
    flash_attn_kernel<<<dim3(32, 64), 128, 0, stream>>>(Qb, Kb, Vtb, Yb);
    gemm_out_kernel<<<dim3(64, 128), 128, 0, stream>>>(Yb, WpTb, out);
}
